// BitLinear_5488968204631
// MI455X (gfx1250) — compile-verified, matrix operand order checked by reading, not run on hardware
//
#include <hip/hip_runtime.h>

typedef int v8i __attribute__((ext_vector_type(8)));
typedef int v4i_b __attribute__((vector_size(16)));   // matches builtin param type

#define IN_F   4096
#define OUT_F  16384
#define M_ROWS 4096   // 2 * 2048
#define BM 128
#define BN 256
#define BK 64

// ---------------------------------------------------------------------------
// gfx1250 async global->LDS path (ASYNCcnt), with fallback to load+ds_store.
// Builtin signature (from clang diagnostic): param1 = AS1 (global) int4*,
// param2 = AS3 (lds) int4*, then imm offset, imm cpol.
// ---------------------------------------------------------------------------
#if defined(__HIP_DEVICE_COMPILE__) && defined(__gfx1250__) && \
    __has_builtin(__builtin_amdgcn_global_load_async_to_lds_b128)
#define USE_ASYNC_LDS 1
#else
#define USE_ASYNC_LDS 0
#endif

#if USE_ASYNC_LDS
__device__ __forceinline__ __attribute__((address_space(1))) v4i_b*
to_global_v4(const void* p) {
    return (__attribute__((address_space(1))) v4i_b*)(unsigned long long)p;
}
__device__ __forceinline__ __attribute__((address_space(3))) v4i_b*
to_lds_v4(const void* p) {
    // AS3 pointers are 32-bit; low 32 bits of the flat LDS-aperture address
    // are exactly the LDS byte address.
    return (__attribute__((address_space(3))) v4i_b*)(unsigned)(unsigned long long)p;
}
#endif

__device__ __forceinline__ void wait_async0() {
#if defined(__HIP_DEVICE_COMPILE__) && defined(__gfx1250__)
#if __has_builtin(__builtin_amdgcn_s_wait_asynccnt)
    __builtin_amdgcn_s_wait_asynccnt(0);
#else
    asm volatile("s_wait_asynccnt 0x0" ::: "memory");
#endif
#endif
}

// ---------------------------------------------------------------------------
// Kernel 1: sum(|weight|) -> ws_sum (atomicAdd, ws_sum memset to 0 beforehand)
// ---------------------------------------------------------------------------
__global__ __launch_bounds__(256) void absmean_kernel(const float* __restrict__ w,
                                                      float* __restrict__ ws_sum,
                                                      long long n4) {
    long long i = (long long)blockIdx.x * blockDim.x + threadIdx.x;
    long long stride = (long long)gridDim.x * blockDim.x;
    const float4* w4 = (const float4*)w;
    float s = 0.0f;
    for (; i < n4; i += stride) {
        float4 v = w4[i];
        s += fabsf(v.x) + fabsf(v.y) + fabsf(v.z) + fabsf(v.w);
    }
    __shared__ float red[256];
    red[threadIdx.x] = s;
    __syncthreads();
    for (int o = 128; o > 0; o >>= 1) {
        if (threadIdx.x < (unsigned)o) red[threadIdx.x] += red[threadIdx.x + o];
        __syncthreads();
    }
    if (threadIdx.x == 0) atomicAdd(ws_sum, red[0]);
}

// ---------------------------------------------------------------------------
// Kernel 2: ternary-quantize weight -> int8 [OUT_F, IN_F]
// ---------------------------------------------------------------------------
__global__ __launch_bounds__(256) void quant_w_kernel(const float* __restrict__ w,
                                                      const float* __restrict__ ws_sum,
                                                      signed char* __restrict__ wq,
                                                      long long n4) {
    long long i = (long long)blockIdx.x * blockDim.x + threadIdx.x;
    if (i >= n4) return;
    float alpha = fmaxf(ws_sum[0] * (1.0f / ((float)OUT_F * (float)IN_F)), 1e-10f);
    float inv = 1.0f / alpha;
    float4 v = ((const float4*)w)[i];
    char4 q;
    q.x = (signed char)(int)fminf(fmaxf(rintf(v.x * inv), -1.0f), 1.0f);
    q.y = (signed char)(int)fminf(fmaxf(rintf(v.y * inv), -1.0f), 1.0f);
    q.z = (signed char)(int)fminf(fmaxf(rintf(v.z * inv), -1.0f), 1.0f);
    q.w = (signed char)(int)fminf(fmaxf(rintf(v.w * inv), -1.0f), 1.0f);
    ((char4*)wq)[i] = q;
}

// ---------------------------------------------------------------------------
// Kernel 3: RMSNorm + per-row int8 activation quant; rowscale[m]=alpha*gamma/127
// One 256-thread block per row (4096 floats/row -> 4 float4 per thread).
// ---------------------------------------------------------------------------
__global__ __launch_bounds__(256) void rmsnorm_quant_kernel(const float* __restrict__ x,
                                                            const float* __restrict__ nw,
                                                            const float* __restrict__ ws_sum,
                                                            signed char* __restrict__ xq,
                                                            float* __restrict__ rowscale) {
    const int m = blockIdx.x;
    const int t = threadIdx.x;
    const float4* xr  = (const float4*)(x + (long long)m * IN_F);
    const float4* nw4 = (const float4*)nw;

    float4 v[4];
    float ssq = 0.0f;
    for (int j = 0; j < 4; ++j) {
        v[j] = xr[j * 256 + t];
        ssq += v[j].x * v[j].x + v[j].y * v[j].y + v[j].z * v[j].z + v[j].w * v[j].w;
    }
    __shared__ float red[256];
    red[t] = ssq;
    __syncthreads();
    for (int o = 128; o > 0; o >>= 1) {
        if (t < o) red[t] += red[t + o];
        __syncthreads();
    }
    float rinv = rsqrtf(red[0] * (1.0f / (float)IN_F) + 1e-6f);
    __syncthreads();

    float mx = 0.0f;
    for (int j = 0; j < 4; ++j) {
        float4 g = nw4[j * 256 + t];
        v[j].x *= rinv * g.x; v[j].y *= rinv * g.y;
        v[j].z *= rinv * g.z; v[j].w *= rinv * g.w;
        mx = fmaxf(mx, fabsf(v[j].x)); mx = fmaxf(mx, fabsf(v[j].y));
        mx = fmaxf(mx, fabsf(v[j].z)); mx = fmaxf(mx, fabsf(v[j].w));
    }
    red[t] = mx;
    __syncthreads();
    for (int o = 128; o > 0; o >>= 1) {
        if (t < o) red[t] = fmaxf(red[t], red[t + o]);
        __syncthreads();
    }
    float gamma = fmaxf(red[0], 1e-10f);
    float qs = 127.0f / gamma;

    char4* xq4 = (char4*)(xq + (long long)m * IN_F);
    for (int j = 0; j < 4; ++j) {
        char4 q;
        q.x = (signed char)(int)fminf(fmaxf(rintf(v[j].x * qs), -128.0f), 127.0f);
        q.y = (signed char)(int)fminf(fmaxf(rintf(v[j].y * qs), -128.0f), 127.0f);
        q.z = (signed char)(int)fminf(fmaxf(rintf(v[j].z * qs), -128.0f), 127.0f);
        q.w = (signed char)(int)fminf(fmaxf(rintf(v[j].w * qs), -128.0f), 127.0f);
        xq4[j * 256 + t] = q;
    }
    if (t == 0) {
        float alpha = fmaxf(ws_sum[0] * (1.0f / ((float)OUT_F * (float)IN_F)), 1e-10f);
        rowscale[m] = alpha * gamma * (1.0f / 127.0f);
    }
}

// ---------------------------------------------------------------------------
// Kernel 4: int8 NT-GEMM via V_WMMA_I32_16X16X64_IU8.
// Block = 256 threads = 8 wave32s, tile 128x256, BK=64 (one WMMA K per stage).
// Wave (wm in 0..1, wn in 0..3) owns a 64x64 sub-tile = 4x4 fragments
// -> 16 v_wmma per k-step per wave. Double-buffered LDS (2*(8KB+16KB)=48KB),
// fed by gfx1250 async global->LDS loads (ASYNCcnt) when available.
// ---------------------------------------------------------------------------
__global__ __launch_bounds__(256) void gemm_iu8_kernel(const signed char* __restrict__ xq,
                                                       const signed char* __restrict__ wq,
                                                       const float* __restrict__ rowscale,
                                                       float* __restrict__ y) {
    __shared__ signed char As[2][BM * BK];
    __shared__ signed char Bs[2][BN * BK];

    const int t    = threadIdx.x;
    const int lane = t & 31;
    const int wave = t >> 5;
    const int wm   = wave & 1;   // 0..1 -> 64-row group
    const int wn   = wave >> 1;  // 0..3 -> 64-col group
    const int bm   = blockIdx.y * BM;
    const int bn   = blockIdx.x * BN;

    // Global->LDS staging mapping:
    //  A tile 128x64B: thread t -> row t>>1, 32 bytes at col (t&1)*32 (2 x b128)
    //  B tile 256x64B: thread t -> row t, full 64 bytes            (4 x b128)
    const int arow = t >> 1;
    const int acol = (t & 1) * 32;
    const signed char* aptr = xq + (long long)(bm + arow) * IN_F + acol;
    const signed char* bptr = wq + (long long)(bn + t) * IN_F;

    const v8i vzero = {0, 0, 0, 0, 0, 0, 0, 0};
    v8i acc[4][4];
    for (int i = 0; i < 4; ++i)
        for (int j = 0; j < 4; ++j)
            acc[i][j] = vzero;

    const int KT = IN_F / BK;  // 64

    // ---- tile staging (async path or reg-staged fallback) ----
#if USE_ASYNC_LDS
    auto issue_tile = [&](int kt, int nb) {
        const signed char* ap = aptr + (long long)kt * BK;
        const signed char* bp = bptr + (long long)kt * BK;
        signed char* la = &As[nb][arow * BK + acol];
        signed char* lb = &Bs[nb][t * BK];
        __builtin_amdgcn_global_load_async_to_lds_b128(to_global_v4(ap),      to_lds_v4(la),      0, 0);
        __builtin_amdgcn_global_load_async_to_lds_b128(to_global_v4(ap + 16), to_lds_v4(la + 16), 0, 0);
        __builtin_amdgcn_global_load_async_to_lds_b128(to_global_v4(bp),      to_lds_v4(lb),      0, 0);
        __builtin_amdgcn_global_load_async_to_lds_b128(to_global_v4(bp + 16), to_lds_v4(lb + 16), 0, 0);
        __builtin_amdgcn_global_load_async_to_lds_b128(to_global_v4(bp + 32), to_lds_v4(lb + 32), 0, 0);
        __builtin_amdgcn_global_load_async_to_lds_b128(to_global_v4(bp + 48), to_lds_v4(lb + 48), 0, 0);
    };
    issue_tile(0, 0);
    wait_async0();
    __syncthreads();
#else
    auto issue_tile = [&](int kt, int nb) {
        const signed char* ap = aptr + (long long)kt * BK;
        const signed char* bp = bptr + (long long)kt * BK;
        int4 a0 = *(const int4*)(ap);
        int4 a1 = *(const int4*)(ap + 16);
        int4 b0 = *(const int4*)(bp);
        int4 b1 = *(const int4*)(bp + 16);
        int4 b2 = *(const int4*)(bp + 32);
        int4 b3 = *(const int4*)(bp + 48);
        *(int4*)&As[nb][arow * BK + acol]      = a0;
        *(int4*)&As[nb][arow * BK + acol + 16] = a1;
        *(int4*)&Bs[nb][t * BK]      = b0;
        *(int4*)&Bs[nb][t * BK + 16] = b1;
        *(int4*)&Bs[nb][t * BK + 32] = b2;
        *(int4*)&Bs[nb][t * BK + 48] = b3;
    };
    issue_tile(0, 0);
    __syncthreads();
#endif

    // ISA 8-bit fragment addressing: lane l holds row (l&15); its four b64
    // chunks sit at K offsets kg+0,+16,+32,+48 with kg = (l>=16)*8.
    const int kg = (lane >> 4) * 8;
    const int fr = lane & 15;

    int buf = 0;
    for (int kt = 0; kt < KT; ++kt) {
        if (kt + 1 < KT) issue_tile(kt + 1, buf ^ 1);

        v8i a[4], b[4];
        for (int i = 0; i < 4; ++i) {
            const signed char* base = &As[buf][(wm * 64 + i * 16 + fr) * BK + kg];
            union { v8i v; unsigned long long q[4]; } u;
            u.q[0] = *(const unsigned long long*)(base);
            u.q[1] = *(const unsigned long long*)(base + 16);
            u.q[2] = *(const unsigned long long*)(base + 32);
            u.q[3] = *(const unsigned long long*)(base + 48);
            a[i] = u.v;
        }
        for (int j = 0; j < 4; ++j) {
            const signed char* base = &Bs[buf][(wn * 64 + j * 16 + fr) * BK + kg];
            union { v8i v; unsigned long long q[4]; } u;
            u.q[0] = *(const unsigned long long*)(base);
            u.q[1] = *(const unsigned long long*)(base + 16);
            u.q[2] = *(const unsigned long long*)(base + 32);
            u.q[3] = *(const unsigned long long*)(base + 48);
            b[j] = u.v;
        }

        for (int i = 0; i < 4; ++i)
            for (int j = 0; j < 4; ++j)
                acc[i][j] = __builtin_amdgcn_wmma_i32_16x16x64_iu8(
                    /*sgn_a=*/true, a[i], /*sgn_b=*/true, b[j], acc[i][j],
                    /*reuse_a=*/false, /*reuse_b=*/false);

#if USE_ASYNC_LDS
        if (kt + 1 < KT) wait_async0();
#endif
        __syncthreads();
        buf ^= 1;
    }

    // Epilogue: C/D layout — VGPR v: lanes 0-15 => M = v, lanes 16-31 => M = 8+v;
    // N = lane & 15.
    const int mb2  = (lane >> 4) * 8;
    const int nloc = lane & 15;
    for (int i = 0; i < 4; ++i) {
        const int mrow_base = bm + wm * 64 + i * 16 + mb2;
        for (int vv = 0; vv < 8; ++vv) {
            const int mrow = mrow_base + vv;
            const float rs = rowscale[mrow];
            for (int j = 0; j < 4; ++j) {
                const int ncol = bn + wn * 64 + j * 16 + nloc;
                y[(long long)mrow * OUT_F + ncol] = (float)acc[i][j][vv] * rs;
            }
        }
    }
}

// ---------------------------------------------------------------------------
// Host launcher
// ---------------------------------------------------------------------------
extern "C" void kernel_launch(void* const* d_in, const int* in_sizes, int n_in,
                              void* d_out, int out_size, void* d_ws, size_t ws_size,
                              hipStream_t stream) {
    (void)in_sizes; (void)n_in; (void)out_size; (void)ws_size;

    const float* x      = (const float*)d_in[0];  // [2,2048,4096]
    const float* weight = (const float*)d_in[1];  // [16384,4096]
    const float* nw     = (const float*)d_in[2];  // [4096]
    float* y            = (float*)d_out;          // [2,2048,16384]

    char* ws = (char*)d_ws;
    float*       ws_sum   = (float*)ws;                                  // 4 B
    float*       rowscale = (float*)(ws + 1024);                         // 16 KB
    signed char* xq       = (signed char*)(ws + (1 << 20));              // 16 MB
    signed char* wq       = (signed char*)(ws + (1 << 20) + (16 << 20)); // 64 MB

    const long long w_n4 = (long long)OUT_F * IN_F / 4;  // 16,777,216

    (void)hipMemsetAsync(ws_sum, 0, sizeof(float), stream);
    absmean_kernel<<<2048, 256, 0, stream>>>(weight, ws_sum, w_n4);
    quant_w_kernel<<<(unsigned)(w_n4 / 256), 256, 0, stream>>>(weight, ws_sum, wq, w_n4);
    rmsnorm_quant_kernel<<<M_ROWS, 256, 0, stream>>>(x, nw, ws_sum, xq, rowscale);

    dim3 grid(OUT_F / BN, M_ROWS / BM);  // (64, 32)
    gemm_iu8_kernel<<<grid, 256, 0, stream>>>(xq, wq, rowscale, y);
}
